// MyAGNNlayer_60241211293934
// MI455X (gfx1250) — matrix operand order repaired and checked by reading, not compiled, not run on hardware
//
#include <hip/hip_runtime.h>
#include <hip/hip_bf16.h>

typedef __attribute__((ext_vector_type(16))) _Float16 v16h;
typedef __attribute__((ext_vector_type(8)))  _Float16 v8h;
typedef __attribute__((ext_vector_type(2)))  _Float16 v2h;
typedef __attribute__((ext_vector_type(8)))  float    v8f;

#define FEAT 64            // feature dim (fixed by reference)
#define ROWB 128           // bytes per f16 row: 64 * 2

// ---------------------------------------------------------------------------
// Kernel 1: zero the output accumulator and the softmax denominators.
// ---------------------------------------------------------------------------
__global__ void agnn_zero_kernel(float* __restrict__ out, long long n_out,
                                 float* __restrict__ denom, int n_den) {
    long long i = (long long)blockIdx.x * blockDim.x + threadIdx.x;
    long long stride = (long long)gridDim.x * blockDim.x;
    for (long long k = i; k < n_out; k += stride) out[k] = 0.0f;
    for (long long k = i; k < n_den; k += stride) denom[k] = 0.0f;
}

// ---------------------------------------------------------------------------
// Kernel 2: L2-normalize each node row, store as f16 row-major [N][64].
// One wave (32 lanes) per node; lane handles 2 features (float2 load).
// ---------------------------------------------------------------------------
__global__ void agnn_norm_kernel(const float* __restrict__ x,
                                 _Float16* __restrict__ xn, int N) {
    int gtid = blockIdx.x * blockDim.x + threadIdx.x;
    int node = gtid >> 5;
    int lane = gtid & 31;
    if (node >= N) return;                      // wave-uniform exit

    float2 v = ((const float2*)(x + (size_t)node * FEAT))[lane];
    float ss = v.x * v.x + v.y * v.y;
    #pragma unroll
    for (int off = 16; off > 0; off >>= 1)
        ss += __shfl_xor(ss, off, 32);

    float nrm = sqrtf(ss);
    float sc  = 1.0f / fmaxf(nrm, 1e-12f);

    v2h h;
    h.x = (_Float16)(v.x * sc);
    h.y = (_Float16)(v.y * sc);
    ((v2h*)(xn + (size_t)node * FEAT))[lane] = h;
}

// ---------------------------------------------------------------------------
// Kernel 3: per-edge attention logits via WMMA.
// One wave handles 16 edges. A = 16 gathered dst rows (16x32 f16, two WMMA
// K-chunks cover K=64); B = 16 gathered src rows as columns (32x16 f16).
// diag(A x B^T-style product) = the 16 edge dot products, f32-accumulated.
// w = exp(beta*cos - |beta|)   (exact softmax shift, numerically safe)
// ---------------------------------------------------------------------------
__global__ void agnn_edge_wmma_kernel(const _Float16* __restrict__ xn,
                                      const int* __restrict__ row,
                                      const int* __restrict__ col,
                                      const float* __restrict__ beta_p,
                                      float* __restrict__ wbuf,
                                      float* __restrict__ denom,
                                      int E) {
    int gtid = blockIdx.x * blockDim.x + threadIdx.x;
    int wave = gtid >> 5;
    int lane = gtid & 31;
    long long eg = (long long)wave * 16;
    if (eg >= E) return;                        // wave-uniform exit

    // --- gather indices for this lane's A-row / B-column (edge m = lane&15)
    int m = lane & 15;
    long long em = eg + m;
    if (em >= E) em = (long long)E - 1;         // clamp (load-only; no EXEC change)
    int r = row[em];
    int s = col[em];

    const char* ra = (const char*)xn + (size_t)r * ROWB;  // dst row (A)
    const char* rb = (const char*)xn + (size_t)s * ROWB;  // src row (B column)

    // A and B both use contiguous K order: lane holds 16 contiguous K,
    // lanes<16 -> K 0..15, lanes>=16 -> K 16..31.
    int hiA = (lane & 16) ? 16 : 0;   // unused; A loads also use hiB
    int hiB = (lane & 16) ? 32 : 0;   // byte offset of this lane's B K-run

    v8f c = {};
    #pragma unroll
    for (int ck = 0; ck < 2; ++ck) {            // two K=32 chunks cover FEAT=64
        int base = ck * 64;                     // 32 halfs = 64 bytes per chunk
        v16h a = *(const v16h*)(ra + base + hiB);
        v16h b = *(const v16h*)(rb + base + hiB);         // 32 contiguous bytes
        c = __builtin_amdgcn_wmma_f32_16x16x32_f16(
                false, a, false, b, (short)0, c, false, false);
    }

    // --- extract diagonal: (i,i) lives at lane i (i<8) / lane 16+i (i>=8),
    //     accumulator VGPR i&7.
    int q = lane & 7;
    float dv = c[0];
    dv = (q == 1) ? c[1] : dv;
    dv = (q == 2) ? c[2] : dv;
    dv = (q == 3) ? c[3] : dv;
    dv = (q == 4) ? c[4] : dv;
    dv = (q == 5) ? c[5] : dv;
    dv = (q == 6) ? c[6] : dv;
    dv = (q == 7) ? c[7] : dv;

    bool valid = (lane < 8) || (lane >= 24);    // lanes holding a diagonal elem
    int  i     = (lane & 7) + ((lane >> 4) << 3);
    long long e = eg + i;
    if (valid && e < E) {
        float beta  = beta_p[0];
        float wv    = __expf(beta * dv - fabsf(beta));   // shifted softmax numerator
        wbuf[e] = wv;
        atomicAdd(&denom[row[e]], wv);
    }
}

// ---------------------------------------------------------------------------
// Kernel 4: attention-weighted aggregation. One wave per edge; 32 lanes x
// float2 cover the 64-wide feature row; 64 f32 atomics into out[row].
// ---------------------------------------------------------------------------
__global__ void agnn_agg_kernel(const float* __restrict__ x,
                                const int* __restrict__ row,
                                const int* __restrict__ col,
                                const float* __restrict__ wbuf,
                                const float* __restrict__ denom,
                                float* __restrict__ out, int E) {
    int gtid = blockIdx.x * blockDim.x + threadIdx.x;
    int edge = gtid >> 5;
    int lane = gtid & 31;
    if (edge >= E) return;                      // wave-uniform exit

    int r = row[edge];
    int s = col[edge];
    float alpha = wbuf[edge] / denom[r];

    float2 xv = ((const float2*)(x + (size_t)s * FEAT))[lane];
    float* dst = out + (size_t)r * FEAT + lane * 2;
    atomicAdd(dst,     alpha * xv.x);
    atomicAdd(dst + 1, alpha * xv.y);
}

// ---------------------------------------------------------------------------
extern "C" void kernel_launch(void* const* d_in, const int* in_sizes, int n_in,
                              void* d_out, int out_size, void* d_ws, size_t ws_size,
                              hipStream_t stream) {
    const float* x    = (const float*)d_in[0];
    const float* beta = (const float*)d_in[1];
    const int*   row  = (const int*)d_in[2];
    const int*   col  = (const int*)d_in[3];
    float*       out  = (float*)d_out;

    int N = in_sizes[0] / FEAT;
    int E = in_sizes[2];

    // workspace layout: [ xn f16 (N*128B) | w (E*4B) | denom (N*4B) ]
    char* ws = (char*)d_ws;
    size_t xnBytes = ((size_t)N * ROWB + 255) & ~(size_t)255;
    size_t wBytes  = ((size_t)E * sizeof(float) + 255) & ~(size_t)255;
    _Float16* xn    = (_Float16*)ws;
    float*    wbuf  = (float*)(ws + xnBytes);
    float*    denom = (float*)(ws + xnBytes + wBytes);

    // 1) zero accumulators (out is poisoned by harness; must re-zero per call)
    agnn_zero_kernel<<<1024, 256, 0, stream>>>(out, (long long)N * FEAT, denom, N);

    // 2) normalize: one wave per node (8 waves / 256-thread block)
    agnn_norm_kernel<<<(N + 7) / 8, 256, 0, stream>>>(x, xn, N);

    // 3) WMMA edge logits + denom: one wave per 16 edges
    int G = (E + 15) / 16;
    agnn_edge_wmma_kernel<<<(G + 7) / 8, 256, 0, stream>>>(
        xn, row, col, beta, wbuf, denom, E);

    // 4) aggregation: one wave per edge
    agnn_agg_kernel<<<(E + 7) / 8, 256, 0, stream>>>(
        x, row, col, wbuf, denom, out, E);
}
